// SASRecBlock_52536039965143
// MI455X (gfx1250) — matrix operand order repaired and checked by reading, not compiled, not run on hardware
//
#include <hip/hip_runtime.h>
#include <hip/hip_bf16.h>

// ---------------------------------------------------------------------------
// Types for CDNA5 WMMA (wave32): v16bf A/B fragments, v8f C/D accumulators.
// ---------------------------------------------------------------------------
typedef __attribute__((ext_vector_type(16))) __bf16        v16bf;
typedef __attribute__((ext_vector_type(8)))  float         v8f;
typedef __attribute__((ext_vector_type(4)))  unsigned int  v4u;
typedef int v4i __attribute__((vector_size(16)));   // matches clang builtin 'V4i'

union Frag { v16bf v; v4u u[2]; };

#define WMMA_BF16(a, b, c) \
    __builtin_amdgcn_wmma_f32_16x16x32_bf16(false, (a), false, (b), (short)0, (c), false, false)

// ---- async global -> LDS path (CDNA5 ASYNCcnt pipe), with fallback --------
#if defined(__has_builtin)
#if __has_builtin(__builtin_amdgcn_global_load_async_to_lds_b128)
#define HAVE_ASYNC_LDS 1
#endif
#endif
#ifndef HAVE_ASYNC_LDS
#define HAVE_ASYNC_LDS 0
#endif

#if HAVE_ASYNC_LDS
__device__ __forceinline__ void async_ld128(const __bf16* g, __bf16* l) {
    __builtin_amdgcn_global_load_async_to_lds_b128(
        (__attribute__((address_space(1))) v4i*)(uintptr_t)g,
        (__attribute__((address_space(3))) v4i*)(unsigned)(uintptr_t)l,
        0, 0);
}
#if __has_builtin(__builtin_amdgcn_s_wait_asynccnt)
#define WAIT_ASYNC() __builtin_amdgcn_s_wait_asynccnt(0)
#else
#define WAIT_ASYNC() asm volatile("s_wait_asynccnt 0x0" ::: "memory")
#endif
#endif

__device__ __forceinline__ __bf16 f2bf(float f) {
    unsigned u = __builtin_bit_cast(unsigned, f);
    u += 0x7FFFu + ((u >> 16) & 1u);               // round-to-nearest-even
    unsigned short s = (unsigned short)(u >> 16);
    return __builtin_bit_cast(__bf16, s);
}

// Problem constants (fixed by the reference)
constexpr int Bsz  = 8;
constexpr int Lseq = 1024;
constexpr int Dm   = 1024;
constexpr int Hn   = 16;
constexpr int HD   = 64;
constexpr int Fd   = 4096;
constexpr int Nrow = Bsz * Lseq;   // 8192

// ---------------------------------------------------------------------------
// Kernel: transpose + f32 -> bf16 conversion of weights.  src[R][C] -> dst[C][R]
// ---------------------------------------------------------------------------
__global__ void tconv(const float* __restrict__ src, __bf16* __restrict__ dst,
                      int R, int C) {
    __shared__ float t[32][33];
    int c0 = blockIdx.x * 32, r0 = blockIdx.y * 32;
    int tx = threadIdx.x, ty = threadIdx.y;
#pragma unroll
    for (int i = 0; i < 4; i++)
        t[ty + 8 * i][tx] = src[(size_t)(r0 + ty + 8 * i) * C + c0 + tx];
    __syncthreads();
#pragma unroll
    for (int i = 0; i < 4; i++)
        dst[(size_t)(c0 + ty + 8 * i) * R + r0 + tx] = f2bf(t[tx][ty + 8 * i]);
}

// ---------------------------------------------------------------------------
// Kernel: row LayerNorm (D=1024).  256 threads = 8 wave32, 4 elems/thread.
// Optionally also writes the raw input converted to bf16 (for x -> K/V GEMMs).
// ---------------------------------------------------------------------------
__global__ void ln_k(const float* __restrict__ in, const float* __restrict__ g,
                     const float* __restrict__ be, float* __restrict__ outF,
                     __bf16* __restrict__ outBF, __bf16* __restrict__ rawBF) {
    int row = blockIdx.x, tid = threadIdx.x;
    const float* xr = in + (size_t)row * Dm;
    float4 xv = ((const float4*)xr)[tid];
    float s  = xv.x + xv.y + xv.z + xv.w;
    float ss = xv.x * xv.x + xv.y * xv.y + xv.z * xv.z + xv.w * xv.w;
#pragma unroll
    for (int o = 1; o < 32; o <<= 1) {
        s  += __shfl_xor(s, o, 32);
        ss += __shfl_xor(ss, o, 32);
    }
    __shared__ float aS[8], aSS[8], bc[2];
    int w = tid >> 5, lane = tid & 31;
    if (lane == 0) { aS[w] = s; aSS[w] = ss; }
    __syncthreads();
    if (tid == 0) {
        float S = 0.f, SS = 0.f;
#pragma unroll
        for (int i = 0; i < 8; i++) { S += aS[i]; SS += aSS[i]; }
        bc[0] = S; bc[1] = SS;
    }
    __syncthreads();
    float mean = bc[0] * (1.0f / Dm);
    float var  = bc[1] * (1.0f / Dm) - mean * mean;
    float rstd = rsqrtf(var + 1e-8f);
    float vals[4] = {xv.x, xv.y, xv.z, xv.w};
#pragma unroll
    for (int i = 0; i < 4; i++) {
        int c = tid * 4 + i;
        float y = (vals[i] - mean) * rstd * g[c] + be[c];
        if (outF) outF[(size_t)row * Dm + c] = y;
        outBF[(size_t)row * Dm + c] = f2bf(y);
        if (rawBF) rawBF[(size_t)row * Dm + c] = f2bf(vals[i]);
    }
}

// ---------------------------------------------------------------------------
// Generic bf16 WMMA GEMM:  C[N=8192][M] = A[N][K] * Bt[M][K]^T  (+ epilogue)
// Block: 256 threads = 8 waves, 128x128 tile; wave = 32x64 (2x4 WMMA tiles).
// Double-buffered LDS fed by async global->LDS loads (ASYNCcnt pipe), K in
// chunks of 32, padded LDS stride (40 halves) to spread bank accesses.
// ---------------------------------------------------------------------------
enum { EPI_QHEAD = 0, EPI_KHEAD, EPI_VTHEAD, EPI_RELU_BF, EPI_FFN2 };

template <int EPI>
__global__ __launch_bounds__(256) void gemm_bf16(
    const __bf16* __restrict__ A, const __bf16* __restrict__ Bt,
    const float* __restrict__ bias, __bf16* __restrict__ obf,
    float* __restrict__ of, const float* __restrict__ resid, int K, int M) {
    __shared__ __attribute__((aligned(16))) __bf16 As[2][128 * 40];
    __shared__ __attribute__((aligned(16))) __bf16 Bs[2][128 * 40];

    int tid = threadIdx.x, w = tid >> 5, lane = tid & 31;
    int ln = lane & 15, lh = lane >> 4;
    int wr = w >> 1, wc = w & 1;
    int rowBase = blockIdx.y * 128, colBase = blockIdx.x * 128;
    int KT = K >> 5;

    v8f acc[2][4] = {};
    int cur = 0;

#if HAVE_ASYNC_LDS
    // ---- async double-buffered pipeline: copy engine writes LDS directly ----
    auto issue = [&](int kt, int buf) {
        int kb = kt * 32;
#pragma unroll
        for (int i = 0; i < 2; i++) {
            int ch = tid + 256 * i, r = ch >> 2, ko = (ch & 3) * 8;
            async_ld128(&A[(size_t)(rowBase + r) * K + kb + ko], &As[buf][r * 40 + ko]);
            async_ld128(&Bt[(size_t)(colBase + r) * K + kb + ko], &Bs[buf][r * 40 + ko]);
        }
    };
    issue(0, 0);
    WAIT_ASYNC();
    __syncthreads();
#else
    // ---- fallback: global -> VGPR -> LDS staging ----
#pragma unroll
    for (int i = 0; i < 2; i++) {
        int ch = tid + 256 * i, r = ch >> 2, ko = (ch & 3) * 8;
        *(v4u*)&As[0][r * 40 + ko] = *(const v4u*)&A[(size_t)(rowBase + r) * K + ko];
        *(v4u*)&Bs[0][r * 40 + ko] = *(const v4u*)&Bt[(size_t)(colBase + r) * K + ko];
    }
    __syncthreads();
#endif

    for (int kt = 0; kt < KT; kt++) {
#if HAVE_ASYNC_LDS
        if (kt + 1 < KT) issue(kt + 1, cur ^ 1);   // overlaps with WMMAs below
#else
        v4u ar[2], br[2];
        if (kt + 1 < KT) {
            int kb = (kt + 1) * 32;
#pragma unroll
            for (int i = 0; i < 2; i++) {
                int ch = tid + 256 * i, r = ch >> 2, ko = (ch & 3) * 8;
                ar[i] = *(const v4u*)&A[(size_t)(rowBase + r) * K + kb + ko];
                br[i] = *(const v4u*)&Bt[(size_t)(colBase + r) * K + kb + ko];
            }
        }
#endif
        if (kt + 2 < KT) {   // L2 prefetch hints (global_prefetch_b8)
            __builtin_prefetch(&A[(size_t)(rowBase + (tid & 127)) * K + (kt + 2) * 32], 0, 0);
            __builtin_prefetch(&Bt[(size_t)(colBase + (tid & 127)) * K + (kt + 2) * 32], 0, 0);
        }

        Frag af[2], bfr[4];
#pragma unroll
        for (int i = 0; i < 2; i++) {
            int r = wr * 32 + i * 16 + ln;
            af[i].u[0] = *(const v4u*)&As[cur][r * 40 + lh * 8];
            af[i].u[1] = *(const v4u*)&As[cur][r * 40 + 16 + lh * 8];
        }
#pragma unroll
        for (int j = 0; j < 4; j++) {
            int c = wc * 64 + j * 16 + ln;
            bfr[j].u[0] = *(const v4u*)&Bs[cur][c * 40 + lh * 8];
            bfr[j].u[1] = *(const v4u*)&Bs[cur][c * 40 + 16 + lh * 8];
        }
#pragma unroll
        for (int i = 0; i < 2; i++)
#pragma unroll
            for (int j = 0; j < 4; j++)
                acc[i][j] = WMMA_BF16(af[i].v, bfr[j].v, acc[i][j]);

#if HAVE_ASYNC_LDS
        if (kt + 1 < KT) WAIT_ASYNC();             // async landed during compute
#else
        if (kt + 1 < KT) {
#pragma unroll
            for (int i = 0; i < 2; i++) {
                int ch = tid + 256 * i, r = ch >> 2, ko = (ch & 3) * 8;
                *(v4u*)&As[cur ^ 1][r * 40 + ko] = ar[i];
                *(v4u*)&Bs[cur ^ 1][r * 40 + ko] = br[i];
            }
        }
#endif
        __syncthreads();
        cur ^= 1;
    }

    // epilogue: C/D layout — VGPR r holds (row r + 8*lh, col ln) of 16x16 tile
#pragma unroll
    for (int i = 0; i < 2; i++) {
#pragma unroll
        for (int j = 0; j < 4; j++) {
            int col = colBase + wc * 64 + j * 16 + ln;
            float bv = bias[col];
#pragma unroll
            for (int r = 0; r < 8; r++) {
                int row = rowBase + wr * 32 + i * 16 + r + 8 * lh;
                float v = acc[i][j][r] + bv;
                if constexpr (EPI == EPI_QHEAD) {
                    int hh = col >> 6, hd = col & 63, b = row >> 10, l = row & 1023;
                    obf[((size_t)((b * Hn + hh) * Lseq + l)) * HD + hd] = f2bf(v * 0.125f);
                } else if constexpr (EPI == EPI_KHEAD) {
                    int hh = col >> 6, hd = col & 63, b = row >> 10, l = row & 1023;
                    obf[((size_t)((b * Hn + hh) * Lseq + l)) * HD + hd] = f2bf(v);
                } else if constexpr (EPI == EPI_VTHEAD) {
                    int hh = col >> 6, hd = col & 63, b = row >> 10, l = row & 1023;
                    obf[((size_t)((b * Hn + hh) * HD + hd)) * Lseq + l] = f2bf(v);
                } else if constexpr (EPI == EPI_RELU_BF) {
                    obf[(size_t)row * M + col] = f2bf(fmaxf(v, 0.0f));
                } else {  // EPI_FFN2: + residual, f32 out
                    of[(size_t)row * Dm + col] = v + resid[(size_t)row * Dm + col];
                }
            }
        }
    }
}

// ---------------------------------------------------------------------------
// Flash-style causal masked attention.  grid = (L/128, B*H), 256 threads.
// Each wave owns 16 query rows; streams keys in chunks of 32.
// No __syncthreads (waves have different causal trip counts) — wave-local LDS
// transpose of P with explicit s_wait_dscnt.
// ---------------------------------------------------------------------------
__global__ __launch_bounds__(256) void attn_k(
    const __bf16* __restrict__ Q, const __bf16* __restrict__ Kc,
    const __bf16* __restrict__ VT, const int* __restrict__ mask,
    const float* __restrict__ qin, float* __restrict__ hout) {
    __shared__ __attribute__((aligned(16))) __bf16 pb[8][512];  // per-wave 16x32

    int tid = threadIdx.x, w = tid >> 5, lane = tid & 31;
    int ln = lane & 15, lh = lane >> 4;
    int bh = blockIdx.y, b = bh >> 4, h = bh & 15;
    int qbase = blockIdx.x * 128 + w * 16;

    // Q A-fragments (16 rows x 64 hd, two K=32 chunks), scale already folded in
    const __bf16* Qrow = Q + ((size_t)bh * Lseq + qbase + ln) * HD;
    Frag qa[2];
#pragma unroll
    for (int c = 0; c < 2; c++) {
        qa[c].u[0] = *(const v4u*)(Qrow + c * 32 + lh * 8);
        qa[c].u[1] = *(const v4u*)(Qrow + c * 32 + 16 + lh * 8);
    }

    v8f O[4] = {};
    float mrow[8], lrow[8];
#pragma unroll
    for (int r = 0; r < 8; r++) { mrow[r] = -3.0e38f; lrow[r] = 0.0f; }
    const int* mb = mask + b * Lseq;

    for (int kb = 0; kb <= qbase + 15; kb += 32) {
        // S = Q K^T for 16 q x 32 keys  (2 key tiles x 2 hd chunks = 4 WMMAs)
        v8f s0 = {}, s1 = {};
        {
            const __bf16* Kp0 = Kc + ((size_t)bh * Lseq + kb + ln) * HD;
            const __bf16* Kp1 = Kc + ((size_t)bh * Lseq + kb + 16 + ln) * HD;
            Frag k0a, k0b, k1a, k1b;
            k0a.u[0] = *(const v4u*)(Kp0 + lh * 8);
            k0a.u[1] = *(const v4u*)(Kp0 + 16 + lh * 8);
            k0b.u[0] = *(const v4u*)(Kp0 + 32 + lh * 8);
            k0b.u[1] = *(const v4u*)(Kp0 + 32 + 16 + lh * 8);
            k1a.u[0] = *(const v4u*)(Kp1 + lh * 8);
            k1a.u[1] = *(const v4u*)(Kp1 + 16 + lh * 8);
            k1b.u[0] = *(const v4u*)(Kp1 + 32 + lh * 8);
            k1b.u[1] = *(const v4u*)(Kp1 + 32 + 16 + lh * 8);
            s0 = WMMA_BF16(qa[0].v, k0a.v, s0);
            s0 = WMMA_BF16(qa[1].v, k0b.v, s0);
            s1 = WMMA_BF16(qa[0].v, k1a.v, s1);
            s1 = WMMA_BF16(qa[1].v, k1b.v, s1);
        }

        int k0 = kb + ln, k1 = k0 + 16;
        int km0 = mb[k0], km1 = mb[k1];
        float p0[8], p1[8], fac[8];
#pragma unroll
        for (int r = 0; r < 8; r++) {
            int q = qbase + r + 8 * lh;
            float a = (km0 && (k0 <= q)) ? s0[r] : -1.0e9f;
            float c = (km1 && (k1 <= q)) ? s1[r] : -1.0e9f;
            float mx = fmaxf(a, c);
#pragma unroll
            for (int o = 1; o < 16; o <<= 1) mx = fmaxf(mx, __shfl_xor(mx, o, 32));
            float mn = fmaxf(mrow[r], mx);
            float e0 = __expf(a - mn), e1 = __expf(c - mn);
            float rs = e0 + e1;
#pragma unroll
            for (int o = 1; o < 16; o <<= 1) rs += __shfl_xor(rs, o, 32);
            fac[r]  = __expf(mrow[r] - mn);
            lrow[r] = lrow[r] * fac[r] + rs;
            mrow[r] = mn;
            p0[r] = e0; p1[r] = e1;
        }
#pragma unroll
        for (int j = 0; j < 4; j++)
#pragma unroll
            for (int r = 0; r < 8; r++) O[j][r] *= fac[r];

        // transpose P (D-layout -> A-layout) via wave-private LDS
#pragma unroll
        for (int r = 0; r < 8; r++) {
            int rr = r + 8 * lh;
            pb[w][rr * 32 + ln]      = f2bf(p0[r]);
            pb[w][rr * 32 + ln + 16] = f2bf(p1[r]);
        }
        asm volatile("s_wait_dscnt 0x0" ::: "memory");
        Frag pa;
        pa.u[0] = *(const v4u*)&pb[w][ln * 32 + lh * 8];
        pa.u[1] = *(const v4u*)&pb[w][ln * 32 + 16 + lh * 8];

        // O += P * V   (V^T layout: contiguous keys per hd column)
#pragma unroll
        for (int j = 0; j < 4; j++) {
            const __bf16* Vp = VT + ((size_t)bh * HD + j * 16 + ln) * Lseq + kb + lh * 8;
            Frag vf;
            vf.u[0] = *(const v4u*)Vp;
            vf.u[1] = *(const v4u*)(Vp + 16);
            O[j] = WMMA_BF16(pa.v, vf.v, O[j]);
        }
    }

    // epilogue: 1/l, query mask, + q_in residual
#pragma unroll
    for (int r = 0; r < 8; r++) {
        int q = qbase + r + 8 * lh;
        float inv = mb[q] ? (1.0f / lrow[r]) : 0.0f;
#pragma unroll
        for (int j = 0; j < 4; j++) {
            int col = h * HD + j * 16 + ln;
            size_t idx = ((size_t)(b * Lseq + q)) * Dm + col;
            hout[idx] = O[j][r] * inv + qin[idx];
        }
    }
}

// ---------------------------------------------------------------------------
// Host-side orchestration
// ---------------------------------------------------------------------------
extern "C" void kernel_launch(void* const* d_in, const int* in_sizes, int n_in,
                              void* d_out, int out_size, void* d_ws, size_t ws_size,
                              hipStream_t stream) {
    (void)in_sizes; (void)n_in; (void)out_size; (void)ws_size;
    const float* x    = (const float*)d_in[0];
    const int*   mask = (const int*)d_in[1];
    const float* wq   = (const float*)d_in[2];
    const float* bq   = (const float*)d_in[3];
    const float* wk   = (const float*)d_in[4];
    const float* bk   = (const float*)d_in[5];
    const float* wv   = (const float*)d_in[6];
    const float* bv   = (const float*)d_in[7];
    const float* w1   = (const float*)d_in[8];
    const float* b1   = (const float*)d_in[9];
    const float* w2   = (const float*)d_in[10];
    const float* b2   = (const float*)d_in[11];
    const float* g1   = (const float*)d_in[12];
    const float* be1  = (const float*)d_in[13];
    const float* g2   = (const float*)d_in[14];
    const float* be2  = (const float*)d_in[15];
    float* out = (float*)d_out;

    char* ws = (char*)d_ws;
    size_t off = 0;
    auto carve = [&](size_t bytes) -> char* {
        char* p = ws + off;
        off = (off + bytes + 255) & ~(size_t)255;
        return p;
    };
    __bf16* xbf   = (__bf16*)carve((size_t)Nrow * Dm * 2);
    float*  qinF  = (float*)carve((size_t)Nrow * Dm * 4);
    __bf16* qinBF = (__bf16*)carve((size_t)Nrow * Dm * 2);
    __bf16* Qb    = (__bf16*)carve((size_t)Nrow * Dm * 2);
    __bf16* Kb    = (__bf16*)carve((size_t)Nrow * Dm * 2);
    __bf16* VTb   = (__bf16*)carve((size_t)Nrow * Dm * 2);
    float*  hF    = (float*)carve((size_t)Nrow * Dm * 4);
    __bf16* ln2bf = (__bf16*)carve((size_t)Nrow * Dm * 2);
    __bf16* act   = (__bf16*)carve((size_t)Nrow * Fd * 2);
    __bf16* wqT   = (__bf16*)carve((size_t)Dm * Dm * 2);
    __bf16* wkT   = (__bf16*)carve((size_t)Dm * Dm * 2);
    __bf16* wvT   = (__bf16*)carve((size_t)Dm * Dm * 2);
    __bf16* w1T   = (__bf16*)carve((size_t)Dm * Fd * 2);
    __bf16* w2T   = (__bf16*)carve((size_t)Fd * Dm * 2);

    dim3 tb(32, 8);
    tconv<<<dim3(Dm / 32, Dm / 32), tb, 0, stream>>>(wq, wqT, Dm, Dm);
    tconv<<<dim3(Dm / 32, Dm / 32), tb, 0, stream>>>(wk, wkT, Dm, Dm);
    tconv<<<dim3(Dm / 32, Dm / 32), tb, 0, stream>>>(wv, wvT, Dm, Dm);
    tconv<<<dim3(Fd / 32, Dm / 32), tb, 0, stream>>>(w1, w1T, Dm, Fd);
    tconv<<<dim3(Dm / 32, Fd / 32), tb, 0, stream>>>(w2, w2T, Fd, Dm);

    ln_k<<<Nrow, 256, 0, stream>>>(x, g1, be1, qinF, qinBF, xbf);

    gemm_bf16<EPI_QHEAD><<<dim3(Dm / 128, Nrow / 128), 256, 0, stream>>>(
        qinBF, wqT, bq, Qb, nullptr, nullptr, Dm, Dm);
    gemm_bf16<EPI_KHEAD><<<dim3(Dm / 128, Nrow / 128), 256, 0, stream>>>(
        xbf, wkT, bk, Kb, nullptr, nullptr, Dm, Dm);
    gemm_bf16<EPI_VTHEAD><<<dim3(Dm / 128, Nrow / 128), 256, 0, stream>>>(
        xbf, wvT, bv, VTb, nullptr, nullptr, Dm, Dm);

    attn_k<<<dim3(Lseq / 128, Bsz * Hn), 256, 0, stream>>>(Qb, Kb, VTb, mask, qinF, hF);

    ln_k<<<Nrow, 256, 0, stream>>>(hF, g2, be2, nullptr, ln2bf, nullptr);

    gemm_bf16<EPI_RELU_BF><<<dim3(Fd / 128, Nrow / 128), 256, 0, stream>>>(
        ln2bf, w1T, b1, act, nullptr, nullptr, Dm, Fd);
    gemm_bf16<EPI_FFN2><<<dim3(Dm / 128, Nrow / 128), 256, 0, stream>>>(
        act, w2T, b2, nullptr, out, hF, Fd, Dm);
}
